// Qwen35TopKRouter_17394617548825
// MI455X (gfx1250) — matrix operand order repaired and checked by reading, not compiled, not run on hardware
//
#include <hip/hip_runtime.h>

// ---------------------------------------------------------------------------
// Qwen3.5 Top-K router for MI455X (gfx1250, wave32).
//   logits = X(T x D) @ W^T(D x E)  -> softmax -> top-8 -> renormalize
// fp32 GEMM emulated with bf16x3 split-precision on V_WMMA_F32_16X16X32_BF16.
// Weight matrix pre-split once into bf16 hi/lo (d_ws), then streamed into a
// double-buffered LDS tile via GLOBAL_LOAD_ASYNC_TO_LDS (ASYNCcnt).
// ---------------------------------------------------------------------------

#define DIM  2048
#define NEXP 128
#define TOPK 8

typedef __attribute__((ext_vector_type(16))) __bf16 bf16x16;
typedef __attribute__((ext_vector_type(8)))  __bf16 bf16x8;
typedef __attribute__((ext_vector_type(4)))  __bf16 bf16x4;
typedef __attribute__((ext_vector_type(8)))  float  f32x8;
typedef __attribute__((ext_vector_type(4)))  int    i32x4;

__device__ __forceinline__ f32x8 wmma_bf16(bf16x16 a, bf16x16 b, f32x8 c) {
  // 8 args: (neg_a, A, neg_b, B, c_mod, C, reuse_a, reuse_b)
  return __builtin_amdgcn_wmma_f32_16x16x32_bf16(false, a, false, b,
                                                 (short)0, c, false, false);
}

// Split fp32 into bf16 hi + bf16 lo residual, writing slot i of two frags.
#define SPLIT_AT(VH, VL, I, VAL)                                   \
  do {                                                             \
    float _x = (VAL);                                              \
    __bf16 _h = (__bf16)_x;                                        \
    (VH)[I] = _h;                                                  \
    (VL)[I] = (__bf16)(_x - (float)_h);                            \
  } while (0)

// ---- Async copy global->LDS (CDNA5), guarded so either toolchain compiles --
#if defined(__gfx1250__) && __has_builtin(__builtin_amdgcn_global_load_async_to_lds_b128)
#define HAVE_ASYNC_LDS 1
typedef __attribute__((address_space(1))) i32x4 g_i32x4;
typedef __attribute__((address_space(3))) i32x4 l_i32x4;
__device__ __forceinline__ void async_cp16(const void* g, void* l) {
  // signature (from hipcc diag): (v4i AS1*, v4i AS3*, imm offset, imm cpol)
  __builtin_amdgcn_global_load_async_to_lds_b128(
      (g_i32x4*)g, (l_i32x4*)l, 0, 0);
}
__device__ __forceinline__ void async_wait0() {
#if __has_builtin(__builtin_amdgcn_s_wait_asynccnt)
  __builtin_amdgcn_s_wait_asynccnt(0);
#else
  asm volatile("s_wait_asynccnt 0x0" ::: "memory");
#endif
}
#else
#define HAVE_ASYNC_LDS 0
#endif

constexpr int KC   = 64;        // K elements staged in LDS per chunk
constexpr int WSTR = KC + 16;   // padded LDS stride (bf16 elems)
constexpr int LSTR = NEXP + 4;  // padded logits stride (floats)

// Stage one K-chunk of the weight matrix into LDS (hi/lo bf16).
// PRESPLIT: source already bf16 hi/lo in global scratch -> pure byte copy.
template <bool PRESPLIT>
__device__ __forceinline__ void stage_chunk(const float* __restrict__ wt,
                                            const __bf16* __restrict__ pwh,
                                            const __bf16* __restrict__ pwl,
                                            __bf16* dsthi, __bf16* dstlo,
                                            int kc) {
  const int e = threadIdx.x;  // 128 threads == 128 experts
  __bf16* dh = dsthi + e * WSTR;
  __bf16* dl = dstlo + e * WSTR;
  if constexpr (PRESPLIT) {
    const __bf16* sh = pwh + (size_t)e * DIM + kc;
    const __bf16* sl = pwl + (size_t)e * DIM + kc;
#if HAVE_ASYNC_LDS
#pragma unroll
    for (int d = 0; d < KC; d += 8) {
      async_cp16(sh + d, dh + d);
      async_cp16(sl + d, dl + d);
    }
#else
#pragma unroll
    for (int d = 0; d < KC; d += 8) {
      *(bf16x8*)(dh + d) = *(const bf16x8*)(sh + d);
      *(bf16x8*)(dl + d) = *(const bf16x8*)(sl + d);
    }
#endif
  } else {
    const float* wr = wt + (size_t)e * DIM + kc;
#pragma unroll
    for (int d = 0; d < KC; d += 8) {
      float4 va = *(const float4*)(wr + d);
      float4 vb = *(const float4*)(wr + d + 4);
      bf16x8 hh, ll;
      SPLIT_AT(hh, ll, 0, va.x);  SPLIT_AT(hh, ll, 1, va.y);
      SPLIT_AT(hh, ll, 2, va.z);  SPLIT_AT(hh, ll, 3, va.w);
      SPLIT_AT(hh, ll, 4, vb.x);  SPLIT_AT(hh, ll, 5, vb.y);
      SPLIT_AT(hh, ll, 6, vb.z);  SPLIT_AT(hh, ll, 7, vb.w);
      *(bf16x8*)(dh + d) = hh;
      *(bf16x8*)(dl + d) = ll;
    }
  }
}

__device__ __forceinline__ void stage_fence() {
#if HAVE_ASYNC_LDS
  async_wait0();  // this wave's async copies landed in LDS
#endif
  __syncthreads();  // all waves' copies visible to everyone
}

// One-time pre-pass: split W (E x D fp32) into bf16 hi + lo residual arrays.
__global__ __launch_bounds__(256) void
presplit_kernel(const float* __restrict__ wt, __bf16* __restrict__ wh,
                __bf16* __restrict__ wl, int n) {
  int i = (blockIdx.x * blockDim.x + threadIdx.x) * 4;
  if (i >= n) return;
  float4 v = *(const float4*)(wt + i);
  bf16x4 h, l;
  SPLIT_AT(h, l, 0, v.x);  SPLIT_AT(h, l, 1, v.y);
  SPLIT_AT(h, l, 2, v.z);  SPLIT_AT(h, l, 3, v.w);
  *(bf16x4*)(wh + i) = h;
  *(bf16x4*)(wl + i) = l;
}

template <bool PRESPLIT>
__global__ __launch_bounds__(128) void
router_topk_kernel(const float* __restrict__ hs,   // (T, D)
                   const float* __restrict__ wt,   // (E, D)
                   const __bf16* __restrict__ pwh, // (E, D) bf16 hi (or null)
                   const __bf16* __restrict__ pwl, // (E, D) bf16 lo (or null)
                   float* __restrict__ outL,       // (T, E) softmax probs
                   float* __restrict__ outW,       // (T, K) renorm weights
                   int*   __restrict__ outI) {     // (T, K) indices (int32)
  __shared__ __align__(16) __bf16 sWhi[2][NEXP * WSTR];
  __shared__ __align__(16) __bf16 sWlo[2][NEXP * WSTR];
  __shared__ __align__(16) float  sLog[64 * LSTR];

  const int lane = threadIdx.x & 31;
  const int wave = threadIdx.x >> 5;   // 0..3, each wave owns 16 tokens
  const int r    = lane & 15;          // row within 16x16 tile
  const int half = lane >> 4;          // K-half selector per WMMA layout
  const int tokBase = blockIdx.x * 64 + wave * 16;

  const float* arow = hs + (size_t)(tokBase + r) * DIM;

  f32x8 acc[8];
  {
    f32x8 z = {0.f, 0.f, 0.f, 0.f, 0.f, 0.f, 0.f, 0.f};
#pragma unroll
    for (int t = 0; t < 8; ++t) acc[t] = z;
  }

  // -------- GEMM: bf16x3, weight chunk double-buffered in LDS -------------
  stage_chunk<PRESPLIT>(wt, pwh, pwl, sWhi[0], sWlo[0], 0);
  stage_fence();

  int cur = 0;
  for (int kc = 0; kc < DIM; kc += KC) {
    const int nxt = cur ^ 1;
    if (kc + KC < DIM)  // prefetch next chunk while we compute this one
      stage_chunk<PRESPLIT>(wt, pwh, pwl, sWhi[nxt], sWlo[nxt], kc + KC);

    __builtin_prefetch(arow + kc + KC + half * 8, 0, 0);  // next A chunk hint

    const __bf16* bhB = sWhi[cur];
    const __bf16* blB = sWlo[cur];
#pragma unroll
    for (int ks = 0; ks < KC / 32; ++ks) {
      // A fragment (contiguous k order, same as B): lane holds row r,
      // K = {base..base+15}, base = half*16.
      const float* ap = arow + kc + ks * 32 + half * 16;
      float4 x0 = *(const float4*)(ap);
      float4 x1 = *(const float4*)(ap + 4);
      float4 x2 = *(const float4*)(ap + 8);
      float4 x3 = *(const float4*)(ap + 12);
      bf16x16 ah, al;
      SPLIT_AT(ah, al, 0,  x0.x); SPLIT_AT(ah, al, 1,  x0.y);
      SPLIT_AT(ah, al, 2,  x0.z); SPLIT_AT(ah, al, 3,  x0.w);
      SPLIT_AT(ah, al, 4,  x1.x); SPLIT_AT(ah, al, 5,  x1.y);
      SPLIT_AT(ah, al, 6,  x1.z); SPLIT_AT(ah, al, 7,  x1.w);
      SPLIT_AT(ah, al, 8,  x2.x); SPLIT_AT(ah, al, 9,  x2.y);
      SPLIT_AT(ah, al, 10, x2.z); SPLIT_AT(ah, al, 11, x2.w);
      SPLIT_AT(ah, al, 12, x3.x); SPLIT_AT(ah, al, 13, x3.y);
      SPLIT_AT(ah, al, 14, x3.z); SPLIT_AT(ah, al, 15, x3.w);

      const int ko = ks * 32 + half * 16;  // B: lane holds 16 consecutive K
#pragma unroll
      for (int t = 0; t < 8; ++t) {
        const __bf16* bhp = bhB + (t * 16 + r) * WSTR + ko;
        const __bf16* blp = blB + (t * 16 + r) * WSTR + ko;
        bf16x16 bh, bl;
        ((bf16x8*)&bh)[0] = *(const bf16x8*)(bhp);
        ((bf16x8*)&bh)[1] = *(const bf16x8*)(bhp + 8);
        ((bf16x8*)&bl)[0] = *(const bf16x8*)(blp);
        ((bf16x8*)&bl)[1] = *(const bf16x8*)(blp + 8);
        acc[t] = wmma_bf16(ah, bh, acc[t]);   // hi*hi
        acc[t] = wmma_bf16(ah, bl, acc[t]);   // hi*lo
        acc[t] = wmma_bf16(al, bh, acc[t]);   // lo*hi
      }
    }
    stage_fence();
    cur = nxt;
  }

  // -------- Dump logits to LDS (C layout: VGPR j -> row j + 8*half) --------
#pragma unroll
  for (int t = 0; t < 8; ++t) {
#pragma unroll
    for (int j = 0; j < 8; ++j) {
      sLog[(wave * 16 + j + half * 8) * LSTR + t * 16 + r] = acc[t][j];
    }
  }
  __syncthreads();

  // ---------------- Softmax + top-8 (one wave per 16 tokens) ---------------
  for (int m = 0; m < 16; ++m) {
    const int tok = tokBase + m;
    float4 v = *(const float4*)&sLog[(wave * 16 + m) * LSTR + lane * 4];

    float mx = fmaxf(fmaxf(v.x, v.y), fmaxf(v.z, v.w));
#pragma unroll
    for (int off = 16; off; off >>= 1)
      mx = fmaxf(mx, __shfl_xor(mx, off, 32));

    float e0 = __expf(v.x - mx), e1 = __expf(v.y - mx);
    float e2 = __expf(v.z - mx), e3 = __expf(v.w - mx);
    float s = e0 + e1 + e2 + e3;
#pragma unroll
    for (int off = 16; off; off >>= 1)
      s += __shfl_xor(s, off, 32);
    float inv = 1.0f / s;

    float p0 = e0 * inv, p1 = e1 * inv, p2 = e2 * inv, p3 = e3 * inv;
    float4 p = make_float4(p0, p1, p2, p3);
    *(float4*)(outL + (size_t)tok * NEXP + lane * 4) = p;

    // Iterative top-8 extraction, lowest-index tie-break (matches lax.top_k).
    float pv0 = p0, pv1 = p1, pv2 = p2, pv3 = p3;
    float myw = 0.f, s8 = 0.f;
    int myi = 0;
#pragma unroll
    for (int k = 0; k < TOPK; ++k) {
      float bv = pv0; int bi = 0;
      if (pv1 > bv) { bv = pv1; bi = 1; }
      if (pv2 > bv) { bv = pv2; bi = 2; }
      if (pv3 > bv) { bv = pv3; bi = 3; }
      int gbi = lane * 4 + bi;
#pragma unroll
      for (int off = 16; off; off >>= 1) {
        float ov = __shfl_xor(bv, off, 32);
        int   oi = __shfl_xor(gbi, off, 32);
        if (ov > bv || (ov == bv && oi < gbi)) { bv = ov; gbi = oi; }
      }
      s8 += bv;
      if (lane == k) { myw = bv; myi = gbi; }
      if ((gbi >> 2) == lane) {       // owner lane retires the winner
        int sl = gbi & 3;
        if      (sl == 0) pv0 = -1.f;
        else if (sl == 1) pv1 = -1.f;
        else if (sl == 2) pv2 = -1.f;
        else              pv3 = -1.f;
      }
    }
    if (lane < TOPK) {
      outW[(size_t)tok * TOPK + lane] = myw / s8;
      outI[(size_t)tok * TOPK + lane] = myi;   // int32 bit pattern
    }
  }
}

extern "C" void kernel_launch(void* const* d_in, const int* in_sizes, int n_in,
                              void* d_out, int out_size, void* d_ws, size_t ws_size,
                              hipStream_t stream) {
  (void)n_in; (void)out_size;
  const float* hs = (const float*)d_in[0];   // (T, D) fp32
  const float* wt = (const float*)d_in[1];   // (E, D) fp32
  const int T = in_sizes[0] / DIM;

  float* outL = (float*)d_out;                       // T*E probs
  float* outW = outL + (size_t)T * NEXP;             // T*K weights
  int*   outI = (int*)(outW + (size_t)T * TOPK);     // T*K indices

  const int    nW     = NEXP * DIM;
  const size_t wbytes = (size_t)nW * 2;              // one bf16 array

  dim3 grid(T / 64), block(128);
  if (ws_size >= 2 * wbytes) {
    // One-time weight split into scratch (deterministic: runs every call).
    __bf16* wh = (__bf16*)d_ws;
    __bf16* wl = (__bf16*)((char*)d_ws + wbytes);
    hipLaunchKernelGGL(presplit_kernel, dim3((nW / 4 + 255) / 256), dim3(256),
                       0, stream, wt, wh, wl, nW);
    hipLaunchKernelGGL((router_topk_kernel<true>), grid, block, 0, stream,
                       hs, wt, wh, wl, outL, outW, outI);
  } else {
    hipLaunchKernelGGL((router_topk_kernel<false>), grid, block, 0, stream,
                       hs, wt, (const __bf16*)nullptr, (const __bf16*)nullptr,
                       outL, outW, outI);
  }
}
